// PtrScorer_12713103197151
// MI455X (gfx1250) — matrix operand order repaired and checked by reading, not compiled, not run on hardware
//
#include <hip/hip_runtime.h>
#include <math.h>

// ---------------- dims (fixed by reference) ----------------
#define NS   4096   // N_SENTS
#define DIN  1024   // D_IN
#define HID_ 1024   // HID
#define KCH  32     // GEMM K chunk (one WMMA depth)

// ---------------- bf16 / WMMA types ----------------
typedef __bf16 bf16_t;
typedef bf16_t v8bf  __attribute__((ext_vector_type(8)));
typedef bf16_t v16bf __attribute__((ext_vector_type(16)));
typedef float  v8f   __attribute__((ext_vector_type(8)));

__device__ __forceinline__ unsigned short f2bf(float f) {
    unsigned int u = __float_as_uint(f);
    unsigned int r = u + 0x7FFFu + ((u >> 16) & 1u);   // round-to-nearest-even
    return (unsigned short)(r >> 16);
}

__device__ __forceinline__ float sigmoidf_(float x) { return 1.0f / (1.0f + __expf(-x)); }

// ---- CDNA5 async global->LDS copy (ASYNCcnt-tracked), 16B per lane ----
__device__ __forceinline__ void async_b128(unsigned lds_off, unsigned long long gaddr) {
    asm volatile("global_load_async_to_lds_b128 %0, %1, off"
                 :: "v"(lds_off), "v"(gaddr) : "memory");
}
__device__ __forceinline__ void wait_async_le3() {
    asm volatile("s_wait_asynccnt 0x3" ::: "memory");
}
__device__ __forceinline__ void wait_async_0() {
    asm volatile("s_wait_asynccnt 0x0" ::: "memory");
}

// ---------------- conversion kernels ----------------
__global__ void k_cvt_bf16(const float* __restrict__ in, unsigned short* __restrict__ out, int n) {
    int i = blockIdx.x * blockDim.x + threadIdx.x;
    if (i < n) out[i] = f2bf(in[i]);
}

// in: [rows, cols] f32 row-major; out: [cols, rows] bf16 (i.e. transposed)
__global__ void k_cvt_bf16_T(const float* __restrict__ in, unsigned short* __restrict__ out,
                             int rows, int cols) {
    int i = blockIdx.x * blockDim.x + threadIdx.x;
    if (i < rows * cols) {
        int r = i / cols, c = i % cols;
        out[(size_t)c * rows + r] = f2bf(in[i]);
    }
}

// ---------------- WMMA GEMM with async-LDS double-buffered staging --------------
// C[M,N] = A[M,K](bf16) * B[K,N], B given as BT[N,K](bf16).
// Block = 256 threads (8 waves) computes a 64(M) x 128(N) macro tile.
// Waves arranged 4(M) x 2(N); each wave owns a 16x64 strip (4 WMMA accumulators).
// Per K-chunk: A chunk 64x32 bf16 (4KB), B chunk 128x32 bf16 (8KB) staged in LDS
// by the async copy engine, double buffered. Pipeline is fully peeled so the
// steady-state loop is branch-free.
__global__ __launch_bounds__(256)
void k_gemm_bf16_wmma(const unsigned short* __restrict__ A16,
                      const unsigned short* __restrict__ BT16,
                      float* __restrict__ C, int M, int Nn, int K)
{
    // per buffer: A = 64*32 = 2048 shorts (4096B), B = 128*32 = 4096 shorts (8192B)
    __shared__ __align__(32) unsigned short smem[2][6144];

    const int tid  = threadIdx.x;
    const int wid  = tid >> 5;
    const int lane = tid & 31;
    const int half = lane >> 4;     // 0: lanes 0-15, 1: lanes 16-31
    const int l16  = lane & 15;

    const int nBlocksN = Nn >> 7;                 // 128-wide N macro tiles
    const int bm = blockIdx.x / nBlocksN;
    const int bn = blockIdx.x % nBlocksN;
    const int rowBase = bm * 64;
    const int colBase = bn * 128;

    // ---- async-copy source mapping for this thread (3 x b128 per chunk) ----
    const int a_row = tid >> 2;                   // 0..63
    const int a_cb  = (tid & 3) * 16;             // byte within 64B row-chunk
    const int b_col = tid >> 1;                   // 0..127
    const int b_cb  = (tid & 1) * 32;             // byte within 64B col-chunk

    const char* Ag  = (const char*)(A16  + (size_t)(rowBase + a_row) * K) + a_cb;
    const char* Bg  = (const char*)(BT16 + (size_t)(colBase + b_col) * K) + b_cb;

    const unsigned lds0 = (unsigned)(uintptr_t)&smem[0][0];   // low 32 bits == LDS byte addr
    const unsigned lds1 = (unsigned)(uintptr_t)&smem[1][0];

    auto issue = [&](int chunk, int buf) {
        const unsigned base = buf ? lds1 : lds0;
        const size_t koff = (size_t)chunk * (KCH * 2);        // bytes along K
        async_b128(base + (unsigned)tid * 16u,
                   (unsigned long long)(uintptr_t)(Ag + koff));
        async_b128(base + 4096u + (unsigned)tid * 32u,
                   (unsigned long long)(uintptr_t)(Bg + koff));
        async_b128(base + 4096u + (unsigned)tid * 32u + 16u,
                   (unsigned long long)(uintptr_t)(Bg + koff + 16));
    };

    const int wm_ = wid & 3;        // wave M sub-tile (0..3)
    const int wn_ = wid >> 2;       // wave N sub-tile (0..1)

    v8f acc0 = {}; v8f acc1 = {}; v8f acc2 = {}; v8f acc3 = {};

    auto compute = [&](int i) {
        const unsigned short* sA = &smem[i & 1][0];
        const unsigned short* sB = &smem[i & 1][2048];

        // A 16x32, contiguous k order used here (B matches): VGPR v holds K = half*16 + 2v (+1), M = lane&15
        const unsigned short* ap = sA + (size_t)(wm_ * 16 + l16) * 32 + half * 16;
        v8bf alo = *reinterpret_cast<const v8bf*>(ap);
        v8bf ahi = *reinterpret_cast<const v8bf*>(ap + 8);
        v16bf a  = __builtin_shufflevector(alo, ahi,
                     0,1,2,3,4,5,6,7,8,9,10,11,12,13,14,15);

        // B 32x16, contiguous k order used here (A matches): VGPR v holds K = half*16 + 2v (+1), N = lane&15
        const unsigned short* bp = sB + (size_t)(wn_ * 64 + l16) * 32 + half * 16;
        v16bf b0 = *reinterpret_cast<const v16bf*>(bp);
        v16bf b1 = *reinterpret_cast<const v16bf*>(bp + 16 * 32);
        v16bf b2 = *reinterpret_cast<const v16bf*>(bp + 32 * 32);
        v16bf b3 = *reinterpret_cast<const v16bf*>(bp + 48 * 32);

        acc0 = __builtin_amdgcn_wmma_f32_16x16x32_bf16(false, a, false, b0, (short)0, acc0, false, false);
        acc1 = __builtin_amdgcn_wmma_f32_16x16x32_bf16(false, a, false, b1, (short)0, acc1, false, false);
        acc2 = __builtin_amdgcn_wmma_f32_16x16x32_bf16(false, a, false, b2, (short)0, acc2, false, false);
        acc3 = __builtin_amdgcn_wmma_f32_16x16x32_bf16(false, a, false, b3, (short)0, acc3, false, false);
    };

    const int nchunks = K / KCH;    // 32 (always >= 2 here)

    // prologue: fill both buffers
    issue(0, 0);
    issue(1, 1);

    // steady state: branch-free
    for (int i = 0; i < nchunks - 2; ++i) {
        wait_async_le3();            // chunk i complete (in-order, chunk i+1 outstanding)
        __syncthreads();             // all waves' copies visible
        compute(i);
        __syncthreads();             // all waves done reading buf[i&1]
        issue(i + 2, i & 1);
    }

    // epilogue: last two chunks, no more issues
    wait_async_le3();                // chunk nchunks-2 complete
    __syncthreads();
    compute(nchunks - 2);
    wait_async_0();                  // chunk nchunks-1 complete
    __syncthreads();
    compute(nchunks - 1);

    // C/D layout: VGPR i -> M = i + 8*half, N = lane&15
    float* Cw = C + (size_t)(rowBase + wm_ * 16 + half * 8) * Nn + colBase + wn_ * 64 + l16;
    #pragma unroll
    for (int i = 0; i < 8; ++i) {
        Cw[(size_t)i * Nn +  0] = acc0[i];
        Cw[(size_t)i * Nn + 16] = acc1[i];
        Cw[(size_t)i * Nn + 32] = acc2[i];
        Cw[(size_t)i * Nn + 48] = acc3[i];
    }
}

// ---------------- LSTM cell: one block per output j ----------------
__global__ __launch_bounds__(128)
void k_lstm(const float* __restrict__ x, const float* __restrict__ h,
            const float* __restrict__ c,
            const float* __restrict__ w_ih, const float* __restrict__ w_hh,
            const float* __restrict__ b_ih, const float* __restrict__ b_hh,
            float* __restrict__ h2, float* __restrict__ c2)
{
    const int j = blockIdx.x;     // 0..HID-1
    const int t = threadIdx.x;    // 0..127
    float p0 = 0.f, p1 = 0.f, p2 = 0.f, p3 = 0.f;
    const size_t r0 = (size_t)(0 * HID_ + j) * DIN;
    const size_t r1 = (size_t)(1 * HID_ + j) * DIN;
    const size_t r2 = (size_t)(2 * HID_ + j) * DIN;
    const size_t r3 = (size_t)(3 * HID_ + j) * DIN;
    for (int k = t; k < DIN; k += 128) {
        float xk = x[k], hk = h[k];
        p0 += w_ih[r0 + k] * xk + w_hh[r0 + k] * hk;
        p1 += w_ih[r1 + k] * xk + w_hh[r1 + k] * hk;
        p2 += w_ih[r2 + k] * xk + w_hh[r2 + k] * hk;
        p3 += w_ih[r3 + k] * xk + w_hh[r3 + k] * hk;
    }
    __shared__ float s[4][128];
    s[0][t] = p0; s[1][t] = p1; s[2][t] = p2; s[3][t] = p3;
    __syncthreads();
    for (int w = 64; w > 0; w >>= 1) {
        if (t < w) {
            s[0][t] += s[0][t + w]; s[1][t] += s[1][t + w];
            s[2][t] += s[2][t + w]; s[3][t] += s[3][t + w];
        }
        __syncthreads();
    }
    if (t == 0) {
        float gi = s[0][0] + b_ih[j]            + b_hh[j];
        float gf = s[1][0] + b_ih[HID_ + j]     + b_hh[HID_ + j];
        float gg = s[2][0] + b_ih[2 * HID_ + j] + b_hh[2 * HID_ + j];
        float go = s[3][0] + b_ih[3 * HID_ + j] + b_hh[3 * HID_ + j];
        float cf = sigmoidf_(gf) * c[j] + sigmoidf_(gi) * tanhf(gg);
        c2[j] = cf;
        h2[j] = sigmoidf_(go) * tanhf(cf);
    }
}

// ---------------- qw[j] = sum_k q[k] * w[k][j] ----------------
__global__ __launch_bounds__(256)
void k_matvec_qw(const float* __restrict__ q, const float* __restrict__ w,
                 float* __restrict__ qw) {
    int j = blockIdx.x * 256 + threadIdx.x;
    float acc = 0.f;
    for (int k = 0; k < HID_; ++k) acc += q[k] * w[(size_t)k * HID_ + j];
    qw[j] = acc;
}

// ---------------- e[n] = tanh(feat[n,:] + qw) . v ----------------
__global__ __launch_bounds__(256)
void k_attn_scores(const float* __restrict__ feat, const float* __restrict__ qw,
                   const float* __restrict__ v, float* __restrict__ e) {
    const int n = blockIdx.x;
    float acc = 0.f;
    for (int j = threadIdx.x; j < HID_; j += 256)
        acc += tanhf(feat[(size_t)n * HID_ + j] + qw[j]) * v[j];
    __shared__ float s[256];
    s[threadIdx.x] = acc;
    __syncthreads();
    for (int w = 128; w > 0; w >>= 1) {
        if (threadIdx.x < w) s[threadIdx.x] += s[threadIdx.x + w];
        __syncthreads();
    }
    if (threadIdx.x == 0) e[n] = s[0];
}

// ---------------- in-place softmax over e[NS] (single block) ----------------
__global__ __launch_bounds__(1024)
void k_softmax(float* __restrict__ e) {
    __shared__ float s[1024];
    const int t = threadIdx.x;
    float m = -3.4e38f;
    for (int n = t; n < NS; n += 1024) m = fmaxf(m, e[n]);
    s[t] = m; __syncthreads();
    for (int w = 512; w > 0; w >>= 1) {
        if (t < w) s[t] = fmaxf(s[t], s[t + w]);
        __syncthreads();
    }
    m = s[0]; __syncthreads();
    float sum = 0.f;
    for (int n = t; n < NS; n += 1024) { float p = __expf(e[n] - m); e[n] = p; sum += p; }
    s[t] = sum; __syncthreads();
    for (int w = 512; w > 0; w >>= 1) {
        if (t < w) s[t] += s[t + w];
        __syncthreads();
    }
    const float inv = 1.0f / s[0];
    for (int n = t; n < NS; n += 1024) e[n] *= inv;
}

// ---------------- context partials: part[ny][j] = sum_{n in chunk ny} p[n]*mem[n][j] ----------------
__global__ __launch_bounds__(256)
void k_ctx_part(const float* __restrict__ p, const float* __restrict__ mem,
                float* __restrict__ part) {
    const int j  = blockIdx.x * 256 + threadIdx.x;   // 0..1023
    const int n0 = blockIdx.y * 256;
    float acc = 0.f;
    for (int n = n0; n < n0 + 256; ++n) {
        __builtin_prefetch(&mem[(size_t)(n + 8) * 1024 + j], 0, 0);  // global_prefetch_b8
        acc += p[n] * mem[(size_t)n * 1024 + j];
    }
    part[(size_t)blockIdx.y * 1024 + j] = acc;
}

__global__ __launch_bounds__(256)
void k_ctx_reduce(const float* __restrict__ part, float* __restrict__ ctx) {
    const int j = blockIdx.x * 256 + threadIdx.x;
    float acc = 0.f;
    #pragma unroll
    for (int i = 0; i < NS / 256; ++i) acc += part[(size_t)i * 1024 + j];
    ctx[j] = acc;
}

// ---------------- score[t] = out . score_w + score_b ----------------
__global__ __launch_bounds__(256)
void k_score(const float* __restrict__ out, const float* __restrict__ sw,
             const float* __restrict__ sb, float* __restrict__ score) {
    float acc = 0.f;
    for (int j = threadIdx.x; j < DIN; j += 256) acc += out[j] * sw[j];
    __shared__ float s[256];
    s[threadIdx.x] = acc;
    __syncthreads();
    for (int w = 128; w > 0; w >>= 1) {
        if (threadIdx.x < w) s[threadIdx.x] += s[threadIdx.x + w];
        __syncthreads();
    }
    if (threadIdx.x == 0) *score = s[0] + sb[0];
}

// ---------------- host orchestration ----------------
extern "C" void kernel_launch(void* const* d_in, const int* in_sizes, int n_in,
                              void* d_out, int out_size, void* d_ws, size_t ws_size,
                              hipStream_t stream) {
    const float* attn_mem = (const float*)d_in[0];
    const float* attn_wm  = (const float*)d_in[1];
    const float* attn_wq  = (const float*)d_in[2];
    const float* attn_v   = (const float*)d_in[3];
    const float* hop_wm   = (const float*)d_in[4];
    const float* hop_wq   = (const float*)d_in[5];
    const float* hop_v    = (const float*)d_in[6];
    const float* init_h   = (const float*)d_in[7];
    const float* init_c   = (const float*)d_in[8];
    const float* init_i   = (const float*)d_in[9];
    const float* w_ih     = (const float*)d_in[10];
    const float* w_hh     = (const float*)d_in[11];
    const float* b_ih     = (const float*)d_in[12];
    const float* b_hh     = (const float*)d_in[13];
    const float* score_w  = (const float*)d_in[14];
    const float* score_b  = (const float*)d_in[15];

    float* scores = (float*)d_out;
    const int T = out_size;   // n_step lives in device memory; out_size == n_step

    // ---- workspace carve-out (256B aligned) ----
    char* ws = (char*)d_ws;
    size_t off = 0;
    auto carve = [&](size_t bytes) -> void* {
        void* p = (void*)(ws + off);
        off += (bytes + 255) & ~(size_t)255;
        return p;
    };
    unsigned short* memb = (unsigned short*)carve((size_t)NS * DIN * 2);   // attn_mem bf16
    unsigned short* wmaT = (unsigned short*)carve((size_t)DIN * HID_ * 2); // attn_wm^T bf16
    unsigned short* wmhT = (unsigned short*)carve((size_t)DIN * HID_ * 2); // hop_wm^T bf16
    float* attn_feat = (float*)carve((size_t)NS * HID_ * 4);
    float* hop_feat  = (float*)carve((size_t)NS * HID_ * 4);
    float* e    = (float*)carve((size_t)NS * 4);
    float* qw   = (float*)carve((size_t)HID_ * 4);
    float* qtmp = (float*)carve((size_t)HID_ * 4);
    float* part = (float*)carve((size_t)(NS / 256) * 1024 * 4);
    float* hA = (float*)carve((size_t)HID_ * 4);
    float* hB = (float*)carve((size_t)HID_ * 4);
    float* cA = (float*)carve((size_t)HID_ * 4);
    float* cB = (float*)carve((size_t)HID_ * 4);
    float* xA = (float*)carve((size_t)DIN * 4);
    float* xB = (float*)carve((size_t)DIN * 4);

    // ---- precompute: bf16 conversions + WMMA GEMMs for feats ----
    k_cvt_bf16  <<<(NS * DIN) / 256, 256, 0, stream>>>(attn_mem, memb, NS * DIN);
    k_cvt_bf16_T<<<(DIN * HID_) / 256, 256, 0, stream>>>(attn_wm, wmaT, DIN, HID_);
    k_cvt_bf16_T<<<(DIN * HID_) / 256, 256, 0, stream>>>(hop_wm,  wmhT, DIN, HID_);
    // (NS/64) * (HID/128) macro tiles = 64 * 8 = 512 blocks
    const int gemmBlocks = (NS / 64) * (HID_ / 128);
    k_gemm_bf16_wmma<<<gemmBlocks, 256, 0, stream>>>(memb, wmaT, attn_feat, NS, HID_, DIN);
    k_gemm_bf16_wmma<<<gemmBlocks, 256, 0, stream>>>(memb, wmhT, hop_feat,  NS, HID_, DIN);

    // ---- sequential scan ----
    const float* x = init_i;
    const float* h = init_h;
    const float* c = init_c;
    for (int t = 0; t < T; ++t) {
        float* hn = (t & 1) ? hB : hA;
        float* cn = (t & 1) ? cB : cA;
        float* xn = (t & 1) ? xB : xA;

        k_lstm<<<HID_, 128, 0, stream>>>(x, h, c, w_ih, w_hh, b_ih, b_hh, hn, cn);

        const float* q = hn;
        for (int hop = 0; hop < 2; ++hop) {
            k_matvec_qw  <<<HID_ / 256, 256, 0, stream>>>(q, hop_wq, qw);
            k_attn_scores<<<NS, 256, 0, stream>>>(hop_feat, qw, hop_v, e);
            k_softmax    <<<1, 1024, 0, stream>>>(e);
            k_ctx_part   <<<dim3(4, NS / 256), 256, 0, stream>>>(e, hop_feat, part);
            k_ctx_reduce <<<HID_ / 256, 256, 0, stream>>>(part, qtmp);
            q = qtmp;
        }
        k_matvec_qw  <<<HID_ / 256, 256, 0, stream>>>(q, attn_wq, qw);
        k_attn_scores<<<NS, 256, 0, stream>>>(attn_feat, qw, attn_v, e);
        k_softmax    <<<1, 1024, 0, stream>>>(e);
        k_ctx_part   <<<dim3(4, NS / 256), 256, 0, stream>>>(e, attn_mem, part);
        k_ctx_reduce <<<DIN / 256, 256, 0, stream>>>(part, xn);

        k_score<<<1, 256, 0, stream>>>(xn, score_w, score_b, scores + t);

        x = xn; h = hn; c = cn;
    }
}
